// DreamAttention_34840774705346
// MI455X (gfx1250) — matrix operand order repaired and checked by reading, not compiled, not run on hardware
//
#include <hip/hip_runtime.h>

typedef __attribute__((ext_vector_type(16))) __bf16 v16bf;
typedef __attribute__((ext_vector_type(8)))  __bf16 v8bf;
typedef __attribute__((ext_vector_type(8)))  float  v8f;

#define T_LEN  2048
#define HID    3584
#define NH     28
#define NKV    4
#define HD     128
#define QDIM   (NH*HD)    // 3584
#define KVDIM  (NKV*HD)   // 512
#define ATT_SCALE 0.08838834764831845f  // 128^-0.5

__device__ __forceinline__ unsigned int pack_bf16(float a, float b) {
  unsigned short ua = __builtin_bit_cast(unsigned short, (__bf16)a);
  unsigned short ub = __builtin_bit_cast(unsigned short, (__bf16)b);
  return (unsigned int)ua | ((unsigned int)ub << 16);
}

// ---------------------------------------------------------------- fp32->bf16
__global__ void cvt_f32_bf16(const float* __restrict__ src,
                             __bf16* __restrict__ dst, int n4) {
  int i = blockIdx.x * blockDim.x + threadIdx.x;
  if (i < n4) {
    float4 v = ((const float4*)src)[i];
    unsigned int lo = pack_bf16(v.x, v.y);
    unsigned int hi = pack_bf16(v.z, v.w);
    ((uint2*)dst)[i] = make_uint2(lo, hi);
  }
}

// ------------------------------------------------- C = A(MxK) * W(NxK)^T + b
// one wave computes a 32(M) x 64(N) tile: each B operand feeds 2 WMMAs,
// 12 b128 loads per 8 WMMAs. fp32 accumulate via WMMA bf16.
__global__ __launch_bounds__(256) void gemm_bf16(
    const __bf16* __restrict__ A, const __bf16* __restrict__ W,
    const float* __restrict__ bias, float* __restrict__ C,
    int M, int N, int K, int tilesN) {
  int wid  = (int)((blockIdx.x * blockDim.x + threadIdx.x) >> 5);
  int lane = threadIdx.x & 31;
  int tm = wid / tilesN, tn = wid - tm * tilesN;
  if (tm * 32 >= M) return;
  int m0 = tm * 32, n0 = tn * 64;
  int col = lane & 15, hi = lane >> 4;

  v8f acc[2][4] = {};
  const __bf16* arow0 = A + (size_t)(m0 + col) * K;
  const __bf16* arow1 = arow0 + (size_t)16 * K;
  const __bf16* w0    = W + (size_t)(n0 + col) * K + hi * 16;

  for (int k0 = 0; k0 < K; k0 += 32) {
    // A operands: lane<16 -> K {0..15}; lane>=16 -> K {16..31}
    v8bf a0lo = *(const v8bf*)(arow0 + k0 + hi * 16);
    v8bf a0hi = *(const v8bf*)(arow0 + k0 + 8 + hi * 16);
    v16bf Aop0 = __builtin_shufflevector(a0lo, a0hi,
        0,1,2,3,4,5,6,7,8,9,10,11,12,13,14,15);
    v8bf a1lo = *(const v8bf*)(arow1 + k0 + hi * 16);
    v8bf a1hi = *(const v8bf*)(arow1 + k0 + 8 + hi * 16);
    v16bf Aop1 = __builtin_shufflevector(a1lo, a1hi,
        0,1,2,3,4,5,6,7,8,9,10,11,12,13,14,15);
    // B operands: lane holds column n, K = hi*16 + e (16 contiguous bf16)
    v16bf B0 = *(const v16bf*)(w0 + k0);
    v16bf B1 = *(const v16bf*)(w0 + (size_t)16 * K + k0);
    v16bf B2 = *(const v16bf*)(w0 + (size_t)32 * K + k0);
    v16bf B3 = *(const v16bf*)(w0 + (size_t)48 * K + k0);
    acc[0][0] = __builtin_amdgcn_wmma_f32_16x16x32_bf16(false, Aop0, false, B0, (short)0, acc[0][0], false, false);
    acc[0][1] = __builtin_amdgcn_wmma_f32_16x16x32_bf16(false, Aop0, false, B1, (short)0, acc[0][1], false, false);
    acc[0][2] = __builtin_amdgcn_wmma_f32_16x16x32_bf16(false, Aop0, false, B2, (short)0, acc[0][2], false, false);
    acc[0][3] = __builtin_amdgcn_wmma_f32_16x16x32_bf16(false, Aop0, false, B3, (short)0, acc[0][3], false, false);
    acc[1][0] = __builtin_amdgcn_wmma_f32_16x16x32_bf16(false, Aop1, false, B0, (short)0, acc[1][0], false, false);
    acc[1][1] = __builtin_amdgcn_wmma_f32_16x16x32_bf16(false, Aop1, false, B1, (short)0, acc[1][1], false, false);
    acc[1][2] = __builtin_amdgcn_wmma_f32_16x16x32_bf16(false, Aop1, false, B2, (short)0, acc[1][2], false, false);
    acc[1][3] = __builtin_amdgcn_wmma_f32_16x16x32_bf16(false, Aop1, false, B3, (short)0, acc[1][3], false, false);
  }

  float bv0 = bias ? bias[n0 + col]      : 0.f;
  float bv1 = bias ? bias[n0 + col + 16] : 0.f;
  float bv2 = bias ? bias[n0 + col + 32] : 0.f;
  float bv3 = bias ? bias[n0 + col + 48] : 0.f;
  float* crow = C + n0 + col;
  #pragma unroll
  for (int g = 0; g < 2; g++) {
    #pragma unroll
    for (int r = 0; r < 8; r++) {
      size_t ro = (size_t)(m0 + g * 16 + hi * 8 + r) * N;
      crow[ro]      = acc[g][0][r] + bv0;
      crow[ro + 16] = acc[g][1][r] + bv1;
      crow[ro + 32] = acc[g][2][r] + bv2;
      crow[ro + 48] = acc[g][3][r] + bv3;
    }
  }
}

// -------------------------------------------------- RoPE + bf16 + V^T layout
__global__ void rope_kernel(const long long* __restrict__ positions,
                            const float* __restrict__ qf,
                            const float* __restrict__ kf,
                            const float* __restrict__ vf,
                            __bf16* __restrict__ qb, __bf16* __restrict__ kb,
                            __bf16* __restrict__ vT) {
  int idx = blockIdx.x * blockDim.x + threadIdx.x;   // T * NH * 64
  if (idx >= T_LEN * NH * (HD / 2)) return;
  int i = idx & 63;
  int h = (idx >> 6) % NH;
  int t = idx / (64 * NH);
  float pos = (float)positions[t];
  // inv_freq = 10000^(-i/64) = exp(-(i/64)*ln(10000))
  float ang = pos * __expf(-(float)i * (1.0f / 64.0f) * 9.210340371976184f);
  float s = __sinf(ang), c = __cosf(ang);

  size_t qbase = (size_t)t * QDIM + h * HD;
  float x1 = qf[qbase + i], x2 = qf[qbase + i + 64];
  qb[qbase + i]      = (__bf16)(x1 * c - x2 * s);
  qb[qbase + i + 64] = (__bf16)(x2 * c + x1 * s);

  if (h < NKV) {
    size_t kbase = (size_t)t * KVDIM + h * HD;
    float k1 = kf[kbase + i], k2 = kf[kbase + i + 64];
    kb[kbase + i]      = (__bf16)(k1 * c - k2 * s);
    kb[kbase + i + 64] = (__bf16)(k2 * c + k1 * s);
    vT[(size_t)(h * HD + i)      * T_LEN + t] = (__bf16)vf[kbase + i];
    vT[(size_t)(h * HD + i + 64) * T_LEN + t] = (__bf16)vf[kbase + i + 64];
  }
}

// ---------------------------------------------- flash attention, 1 wave/task
// task = (head, 16-query block). Computes S^T = K*Q^T tiles (lane = q column),
// streaming softmax per lane, O^T = V^T * P^T accumulated over 32-key steps.
__global__ __launch_bounds__(256) void attn_flash(
    const __bf16* __restrict__ Q, const __bf16* __restrict__ Kmat,
    const __bf16* __restrict__ Vt, __bf16* __restrict__ Ob) {
  int wid  = (int)((blockIdx.x * blockDim.x + threadIdx.x) >> 5);
  int lane = threadIdx.x & 31;
  int h = wid >> 7;           // 128 query blocks per head
  int qblk = wid & 127;
  if (h >= NH) return;
  int t0 = qblk * 16;
  int col = lane & 15, hi = lane >> 4;
  int hkv = h / (NH / NKV);

  // resident Q as B operand (N = query row, K = d), 4 chunks of K=32
  const __bf16* qrow = Q + (size_t)(t0 + col) * QDIM + h * HD;
  v16bf Qop[4];
  #pragma unroll
  for (int c = 0; c < 4; c++)
    Qop[c] = *(const v16bf*)(qrow + c * 32 + hi * 16);

  v8f O[8] = {};              // O^T accumulators, 8 d-tiles of 16
  float mrun = -3.0e38f, ssum = 0.f;
  const __bf16* kbase = Kmat + hkv * HD + (size_t)col * KVDIM;
  const __bf16* vbase = Vt + (size_t)hkv * HD * T_LEN;

  for (int s0 = 0; s0 < T_LEN; s0 += 32) {
    v8f S0 = {}, S1 = {};     // S^T tiles: kpos s0..15 / s0+16..31
    const __bf16* kr0 = kbase + (size_t)s0 * KVDIM;
    const __bf16* kr1 = kr0 + (size_t)16 * KVDIM;
    #pragma unroll
    for (int c = 0; c < 4; c++) {
      v8bf a0 = *(const v8bf*)(kr0 + c * 32 + hi * 16);
      v8bf a1 = *(const v8bf*)(kr0 + c * 32 + 8 + hi * 16);
      v16bf Ak0 = __builtin_shufflevector(a0, a1,
          0,1,2,3,4,5,6,7,8,9,10,11,12,13,14,15);
      S0 = __builtin_amdgcn_wmma_f32_16x16x32_bf16(false, Ak0, false, Qop[c], (short)0, S0, false, false);
      v8bf b0 = *(const v8bf*)(kr1 + c * 32 + hi * 16);
      v8bf b1 = *(const v8bf*)(kr1 + c * 32 + 8 + hi * 16);
      v16bf Ak1 = __builtin_shufflevector(b0, b1,
          0,1,2,3,4,5,6,7,8,9,10,11,12,13,14,15);
      S1 = __builtin_amdgcn_wmma_f32_16x16x32_bf16(false, Ak1, false, Qop[c], (short)0, S1, false, false);
    }
    // online softmax; each lane holds 16 of 32 kpos for its q column,
    // partner (lane^16) holds the other 16
    float lmax = -3.0e38f;
    #pragma unroll
    for (int r = 0; r < 8; r++) {
      S0[r] *= ATT_SCALE; S1[r] *= ATT_SCALE;
      lmax = fmaxf(lmax, fmaxf(S0[r], S1[r]));
    }
    lmax = fmaxf(lmax, __shfl_xor(lmax, 16, 32));
    float mnew  = fmaxf(mrun, lmax);
    float alpha = __expf(mrun - mnew);
    float lsum = 0.f;
    #pragma unroll
    for (int r = 0; r < 8; r++) {
      float p0 = __expf(S0[r] - mnew); S0[r] = p0; lsum += p0;
      float p1 = __expf(S1[r] - mnew); S1[r] = p1; lsum += p1;
    }
    lsum += __shfl_xor(lsum, 16, 32);
    ssum = ssum * alpha + lsum;
    mrun = mnew;

    // assemble P^T as 32x16 B operand (lane = q col, K = s):
    // lane<16 needs s {0..7 own S0, 8..15 partner S0};
    // lane>=16 needs s {16..23 partner S1, 24..31 own S1}
    unsigned pk0[4], pk1[4];
    #pragma unroll
    for (int j = 0; j < 4; j++) {
      pk0[j] = pack_bf16(S0[2 * j], S0[2 * j + 1]);
      pk1[j] = pack_bf16(S1[2 * j], S1[2 * j + 1]);
    }
    union { v16bf v; unsigned u[8]; } Bop;
    #pragma unroll
    for (int j = 0; j < 4; j++) {
      unsigned sw0 = __shfl_xor(pk0[j], 16, 32);
      unsigned sw1 = __shfl_xor(pk1[j], 16, 32);
      Bop.u[j]     = hi ? sw1    : pk0[j];
      Bop.u[4 + j] = hi ? pk1[j] : sw0;
    }

    // O^T += V^T(tile) * P^T   with running rescale
    #pragma unroll
    for (int dt = 0; dt < 8; dt++) {
      #pragma unroll
      for (int r = 0; r < 8; r++) O[dt][r] *= alpha;
      const __bf16* vrow = vbase + (size_t)(dt * 16 + col) * T_LEN + s0 + hi * 16;
      v8bf v0 = *(const v8bf*)vrow;
      v8bf v1 = *(const v8bf*)(vrow + 8);
      v16bf Av = __builtin_shufflevector(v0, v1,
          0,1,2,3,4,5,6,7,8,9,10,11,12,13,14,15);
      O[dt] = __builtin_amdgcn_wmma_f32_16x16x32_bf16(false, Av, false, Bop.v, (short)0, O[dt], false, false);
    }
  }

  float inv = 1.0f / ssum;
  // O^T tile layout: lane holds q row t0+col, VGPR r + hi*8 -> d offset
  __bf16* orow = Ob + (size_t)(t0 + col) * QDIM + h * HD + hi * 8;
  #pragma unroll
  for (int dt = 0; dt < 8; dt++) {
    #pragma unroll
    for (int j = 0; j < 4; j++) {
      unsigned pk = pack_bf16(O[dt][2 * j] * inv, O[dt][2 * j + 1] * inv);
      *(unsigned*)(orow + dt * 16 + 2 * j) = pk;
    }
  }
}

// ---------------------------------------------------------------------------
extern "C" void kernel_launch(void* const* d_in, const int* in_sizes, int n_in,
                              void* d_out, int out_size, void* d_ws, size_t ws_size,
                              hipStream_t stream) {
  const long long* positions = (const long long*)d_in[0];
  const float* hs = (const float*)d_in[1];
  const float* wq = (const float*)d_in[2];
  const float* bq = (const float*)d_in[3];
  const float* wk = (const float*)d_in[4];
  const float* bk = (const float*)d_in[5];
  const float* wv = (const float*)d_in[6];
  const float* bv = (const float*)d_in[7];
  const float* wo = (const float*)d_in[8];
  float* out = (float*)d_out;

  char* ws = (char*)d_ws;
  size_t off = 0;
  auto alloc = [&](size_t bytes) {
    void* p = ws + off;
    off = (off + bytes + 255) & ~(size_t)255;
    return p;
  };
  __bf16* hsb  = (__bf16*)alloc((size_t)T_LEN * HID   * 2);
  __bf16* wqb  = (__bf16*)alloc((size_t)QDIM  * HID   * 2);
  __bf16* wkb  = (__bf16*)alloc((size_t)KVDIM * HID   * 2);
  __bf16* wvb  = (__bf16*)alloc((size_t)KVDIM * HID   * 2);
  __bf16* wob  = (__bf16*)alloc((size_t)HID   * QDIM  * 2);
  float*  qf   = (float*) alloc((size_t)T_LEN * QDIM  * 4);
  float*  kf   = (float*) alloc((size_t)T_LEN * KVDIM * 4);
  float*  vf   = (float*) alloc((size_t)T_LEN * KVDIM * 4);
  __bf16* qb2  = (__bf16*)alloc((size_t)T_LEN * QDIM  * 2);
  __bf16* kb2  = (__bf16*)alloc((size_t)T_LEN * KVDIM * 2);
  __bf16* vTb  = (__bf16*)alloc((size_t)KVDIM * T_LEN * 2);
  __bf16* attb = (__bf16*)alloc((size_t)T_LEN * QDIM  * 2);

  auto cvt = [&](const float* s, __bf16* d, size_t n) {
    int n4 = (int)(n / 4);
    cvt_f32_bf16<<<(n4 + 255) / 256, 256, 0, stream>>>(s, d, n4);
  };
  cvt(hs, hsb, (size_t)T_LEN * HID);
  cvt(wq, wqb, (size_t)QDIM  * HID);
  cvt(wk, wkb, (size_t)KVDIM * HID);
  cvt(wv, wvb, (size_t)KVDIM * HID);
  cvt(wo, wob, (size_t)HID   * QDIM);

  auto gemm = [&](const __bf16* A, const __bf16* W, const float* bias,
                  float* C, int M, int N, int K) {
    int tilesN = N / 64;
    int waves  = (M / 32) * tilesN;
    gemm_bf16<<<(waves * 32 + 255) / 256, 256, 0, stream>>>(A, W, bias, C, M, N, K, tilesN);
  };
  gemm(hsb, wqb, bq, qf, T_LEN, QDIM,  HID);
  gemm(hsb, wkb, bk, kf, T_LEN, KVDIM, HID);
  gemm(hsb, wvb, bv, vf, T_LEN, KVDIM, HID);

  int nrope = T_LEN * NH * (HD / 2);
  rope_kernel<<<(nrope + 255) / 256, 256, 0, stream>>>(positions, qf, kf, vf, qb2, kb2, vTb);

  int awaves = NH * (T_LEN / 16);
  attn_flash<<<(awaves * 32 + 255) / 256, 256, 0, stream>>>(qb2, kb2, vTb, attb);

  gemm(attb, wob, nullptr, out, T_LEN, HID, QDIM);
}
